// NetGlobal_39865886442004
// MI455X (gfx1250) — matrix operand order repaired and checked by reading, not compiled, not run on hardware
//
#include <hip/hip_runtime.h>
#include <hip/hip_bf16.h>
#include <stdint.h>

// ---------------- problem constants (from reference) ----------------
#define BGR   128          // graphs
#define NPG   512          // nodes per graph
#define NFEAT 128
#define NHID  128
#define NCLS  10
#define DEG   16
#define NTOT  (BGR * NPG)          // 65536 nodes
#define ETOT  (NTOT * DEG)         // 1048576 edges
#define KKEEP 256                  // top-k per graph
#define HCAT  (3 * NHID)           // 384
#define RSIZE (6 * NHID)           // 768

typedef __attribute__((ext_vector_type(16))) __bf16 v16bf;
typedef __attribute__((ext_vector_type(8)))  __bf16 v8bf;
typedef __attribute__((ext_vector_type(8)))  float  v8f;

// ---------------- degree / dis = rsqrt(deg) ----------------
__global__ void k_init_deg(float* deg) {
    int i = blockIdx.x * blockDim.x + threadIdx.x;
    if (i < NTOT) deg[i] = 1.0f;                // self loop
}
__global__ void k_count_deg(const int* __restrict__ dst, float* deg) {
    int e = blockIdx.x * blockDim.x + threadIdx.x;
    if (e < ETOT) atomicAdd(&deg[dst[e]], 1.0f);
}
__global__ void k_finish_dis(float* dis) {
    int i = blockIdx.x * blockDim.x + threadIdx.x;
    if (i < NTOT) dis[i] = rsqrtf(dis[i]);
}

// ---------------- f32 -> bf16 elementwise ----------------
__global__ void k_cvt_bf16(const float* __restrict__ in, __bf16* __restrict__ out, int n) {
    int i = blockIdx.x * blockDim.x + threadIdx.x;
    if (i < n) out[i] = (__bf16)in[i];
}

// ---------------- weight transpose + f32 -> bf16 ----------------
// W: [Kdim, Cout] row-major (as in reference)  ->  Wt: [Cout, Kdim] bf16
__global__ void k_transpose_bf16(const float* __restrict__ W, __bf16* __restrict__ Wt,
                                 int Kdim, int Cout) {
    int i = blockIdx.x * blockDim.x + threadIdx.x;
    if (i >= Kdim * Cout) return;
    int co = i / Kdim;
    int k  = i - co * Kdim;
    Wt[i] = (__bf16)W[(size_t)k * Cout + co];
}

// ---------------- WMMA GEMM ----------------
// out[M, ncolblk*CT*16] = A[M,KDIM](bf16) * Wt^T + bias.  Wt is [Cout, KDIM]
// bf16 row-major (W transposed).  blockIdx.y selects a CT*16-wide column
// block; the block stages those Wt rows (KC at a time) into padded LDS
// (bank-conflict-free ds_load_b128).  Each wave computes MT 16-row tiles x
// CT*16 cols; each staged B tile feeds MT back-to-back WMMAs.  A is bf16 in
// memory: two b128 loads per tile per K-step, no cvt VALU, so no
// WMMA->VALU WAR hazard NOPs.  Budget (MT=2,CT=4): acc 64 + A 16 + B 8.
template <int CT, int KDIM, int MT>
__global__ void k_wmma_gemm(const __bf16* __restrict__ A, int lda,
                            const __bf16* __restrict__ Wt,
                            const float* __restrict__ bias,
                            float* __restrict__ out, __bf16* __restrict__ outbf,
                            int ldo, int rowTiles, int relu) {
    constexpr int KC  = 128;        // K columns staged per chunk (KDIM % KC == 0)
    constexpr int LDB = KC + 8;     // padded LDS row stride: lane idx -> bank 4*idx
    __shared__ __bf16 lb[CT * 16 * LDB];

    const int tid  = threadIdx.x;
    const int wave = tid >> 5;
    const int lane = tid & 31;
    const int tileBase = (blockIdx.x * (blockDim.x >> 5) + wave) * MT;
    const bool active = tileBase < rowTiles;    // wave-uniform
    const int half = lane >> 4;                 // 0: lanes 0-15, 1: lanes 16-31
    const int idx  = lane & 15;
    const int colBase = blockIdx.y * (CT * 16);
    const __bf16* WtB = Wt + (size_t)colBase * KDIM;

    v8f acc[MT][CT];
#pragma unroll
    for (int mt = 0; mt < MT; ++mt)
#pragma unroll
        for (int c = 0; c < CT; ++c)
#pragma unroll
            for (int r = 0; r < 8; ++r) acc[mt][c][r] = 0.0f;

    const __bf16* arow[MT];
#pragma unroll
    for (int mt = 0; mt < MT; ++mt)
        arow[mt] = A + (size_t)((tileBase + mt) * 16 + idx) * lda + half * 16;

    for (int kc = 0; kc < KDIM; kc += KC) {
        // ---- cooperative stage of WtB[:, kc:kc+KC) into LDS ----
        if (kc > 0) __syncthreads();            // prior chunk readers done
        for (int id = tid; id < CT * 16 * (KC / 8); id += blockDim.x) {
            const int rowi = id / (KC / 8);
            const int c16  = id - rowi * (KC / 8);
            const uint4 v = *(const uint4*)(WtB + (size_t)rowi * KDIM + kc + c16 * 8);
            *(uint4*)(lb + rowi * LDB + c16 * 8) = v;
        }
        __syncthreads();

        if (active) {
#pragma unroll
            for (int k0 = 0; k0 < KC; k0 += 32) {
                // A tiles (bf16): row (tile*16+idx), chunks {base..+8},{base+8..+8}
                v16bf a[MT];
#pragma unroll
                for (int mt = 0; mt < MT; ++mt) {
                    const __bf16* ap = arow[mt] + kc + k0;
                    __builtin_prefetch(ap + 64, 0, 3);      // global_prefetch_b8
                    const v8bf lo = *(const v8bf*)(ap);
                    const v8bf hi = *(const v8bf*)(ap + 8);
                    a[mt] = __builtin_shufflevector(lo, hi,
                        0, 1, 2, 3, 4, 5, 6, 7, 8, 9, 10, 11, 12, 13, 14, 15);
                }

                // B tiles from LDS; each feeds MT WMMAs
#pragma unroll
                for (int c = 0; c < CT; ++c) {
                    const v16bf b =
                        *(const v16bf*)(lb + (c * 16 + idx) * LDB + k0 + half * 16);
#pragma unroll
                    for (int mt = 0; mt < MT; ++mt)
                        acc[mt][c] = __builtin_amdgcn_wmma_f32_16x16x32_bf16(
                            false, a[mt], false, b, (short)0, acc[mt][c], false, false);
                }
            }
        }
    }

    if (active) {
        // D layout: VGPR r -> M = row0 + half*8 + r, N = colBase + c*16 + idx
#pragma unroll
        for (int mt = 0; mt < MT; ++mt) {
            const int row0 = (tileBase + mt) * 16;
#pragma unroll
            for (int c = 0; c < CT; ++c) {
                const int n = colBase + c * 16 + idx;
                const float bv = bias ? bias[n] : 0.0f;
#pragma unroll
                for (int r = 0; r < 8; ++r) {
                    const int m = row0 + half * 8 + r;
                    float v = acc[mt][c][r] + bv;
                    if (relu) v = fmaxf(v, 0.0f);
                    out[(size_t)m * ldo + n] = v;
                    if (outbf) outbf[(size_t)m * ldo + n] = (__bf16)v;
                }
            }
        }
    }
}

// ---------------- GCN edge aggregation (per-graph, 256 KB LDS) ----------------
// v = sum_{e:dst=n} h[src]*dis[src]*dis[dst] + h[n]*dis[n]^2 + b[f]; relu.
// Writes f32 into hcat (col block) and bf16 into hb (dense [N,NHID], next GEMM's A).
__global__ void k_aggregate(const float* __restrict__ h,
                            const int* __restrict__ src, const int* __restrict__ dst,
                            const float* __restrict__ dis, const float* __restrict__ bias,
                            float* __restrict__ hcat, __bf16* __restrict__ hb,
                            int colOff) {
    extern __shared__ float lds[];              // NPG * NHID floats = 256 KB
    const int g    = blockIdx.x;
    const int tid  = threadIdx.x;
    const int wave = tid >> 5, lane = tid & 31, nw = blockDim.x >> 5;

    for (int i = tid; i < NPG * NHID; i += blockDim.x) lds[i] = 0.0f;
    __syncthreads();

    const int e0 = g * NPG * DEG;
    for (int e = e0 + wave; e < e0 + NPG * DEG; e += nw) {
        const int s = src[e];
        const int d = dst[e];
        const float norm = dis[s] * dis[d];
        const int dl = d - g * NPG;
        const float* hs = h + (size_t)s * NHID;
#pragma unroll
        for (int j = 0; j < 4; ++j) {
            const int f = lane + 32 * j;
            atomicAdd(&lds[dl * NHID + f], hs[f] * norm);   // ds_add_f32
        }
    }
    __syncthreads();

    const int base = g * NPG;
    for (int i = tid; i < NPG * NHID; i += blockDim.x) {
        const int nl = i >> 7, f = i & (NHID - 1);
        const int n = base + nl;
        const float d2 = dis[n] * dis[n];
        float v = lds[i] + h[(size_t)n * NHID + f] * d2 + bias[f];
        v = fmaxf(v, 0.0f);
        hcat[(size_t)n * HCAT + colOff + f] = v;
        hb[(size_t)n * NHID + f] = (__bf16)v;
    }
}

// ---------------- score GCNConv: matvec then scalar aggregation ----------------
__global__ void k_score_matvec(const float* __restrict__ hcat,
                               const float* __restrict__ wsv,   // [384]
                               float* __restrict__ hs) {
    const int wv = (blockIdx.x * blockDim.x + threadIdx.x) >> 5;
    const int lane = threadIdx.x & 31;
    if (wv >= NTOT) return;
    const float* rowp = hcat + (size_t)wv * HCAT;
    float a = 0.0f;
    for (int j = lane; j < HCAT; j += 32) a += rowp[j] * wsv[j];
#pragma unroll
    for (int off = 16; off > 0; off >>= 1) a += __shfl_xor(a, off, 32);
    if (lane == 0) hs[wv] = a;
}

__global__ void k_score_aggregate(const float* __restrict__ hs,
                                  const int* __restrict__ src, const int* __restrict__ dst,
                                  const float* __restrict__ dis, const float* __restrict__ bs,
                                  float* __restrict__ score) {
    __shared__ float lds[NPG];
    const int g = blockIdx.x, tid = threadIdx.x;
    for (int i = tid; i < NPG; i += blockDim.x) lds[i] = 0.0f;
    __syncthreads();
    const int e0 = g * NPG * DEG;
    for (int e = e0 + tid; e < e0 + NPG * DEG; e += blockDim.x) {
        const int s = src[e], d = dst[e];
        atomicAdd(&lds[d - g * NPG], hs[s] * dis[s] * dis[d]);
    }
    __syncthreads();
    for (int i = tid; i < NPG; i += blockDim.x) {
        const int n = g * NPG + i;
        score[n] = lds[i] + hs[n] * dis[n] * dis[n] + bs[0];
    }
}

// ---------------- SAGPool top-k + gated max/mean readout ----------------
__global__ void k_sagpool_readout(const float* __restrict__ hcat,
                                  const float* __restrict__ score,
                                  float* __restrict__ readout) {
    __shared__ float s[NPG];
    __shared__ float gate[NPG];
    __shared__ int   sel[NPG];
    const int g = blockIdx.x, tid = threadIdx.x;

    for (int i = tid; i < NPG; i += blockDim.x) s[i] = score[g * NPG + i];
    __syncthreads();
    for (int i = tid; i < NPG; i += blockDim.x) {
        const float si = s[i];
        int rank = 0;
        for (int j = 0; j < NPG; ++j) {
            const float sj = s[j];
            rank += (sj > si) || (sj == si && j < i);   // stable top-k rank
        }
        sel[i]  = (rank < KKEEP);
        gate[i] = tanhf(si);
    }
    __syncthreads();

    for (int f = tid; f < HCAT; f += blockDim.x) {
        float mx = -3.402823466e38f, sum = 0.0f;
        for (int nl = 0; nl < NPG; ++nl) {
            if (sel[nl]) {                                // uniform branch
                const float v = hcat[(size_t)(g * NPG + nl) * HCAT + f] * gate[nl];
                mx = fmaxf(mx, v);
                sum += v;
            }
        }
        readout[(size_t)g * RSIZE + f]          = mx;
        readout[(size_t)g * RSIZE + HCAT + f]   = sum * (1.0f / KKEEP);
    }
}

// ---------------- final logits + log_softmax ----------------
__global__ void k_logits_lsm(const float* __restrict__ z2,      // [B, 128]
                             const float* __restrict__ Wl3,     // [128, 10]
                             const float* __restrict__ bl3,
                             float* __restrict__ out) {         // [B, 10]
    const int row = blockIdx.x * blockDim.x + threadIdx.x;
    if (row >= BGR) return;
    float acc[NCLS];
#pragma unroll
    for (int c = 0; c < NCLS; ++c) acc[c] = bl3[c];
    const float* zp = z2 + (size_t)row * NHID;
    for (int j = 0; j < NHID; ++j) {
        const float v = zp[j];
#pragma unroll
        for (int c = 0; c < NCLS; ++c) acc[c] += v * Wl3[j * NCLS + c];
    }
    float m = acc[0];
#pragma unroll
    for (int c = 1; c < NCLS; ++c) m = fmaxf(m, acc[c]);
    float sum = 0.0f;
#pragma unroll
    for (int c = 0; c < NCLS; ++c) sum += __expf(acc[c] - m);
    const float lse = m + __logf(sum);
#pragma unroll
    for (int c = 0; c < NCLS; ++c) out[row * NCLS + c] = acc[c] - lse;
}

// ---------------- host side ----------------
extern "C" void kernel_launch(void* const* d_in, const int* in_sizes, int n_in,
                              void* d_out, int out_size, void* d_ws, size_t ws_size,
                              hipStream_t stream) {
    const float* x   = (const float*)d_in[0];
    const int*   ei  = (const int*)d_in[1];
    const int*   src = ei;
    const int*   dst = ei + ETOT;
    const float* W1  = (const float*)d_in[2];   const float* b1  = (const float*)d_in[3];
    const float* W2  = (const float*)d_in[4];   const float* b2  = (const float*)d_in[5];
    const float* W3  = (const float*)d_in[6];   const float* b3  = (const float*)d_in[7];
    const float* Ws  = (const float*)d_in[8];   const float* bs  = (const float*)d_in[9];
    const float* Wl1 = (const float*)d_in[10];  const float* bl1 = (const float*)d_in[11];
    const float* Wl2 = (const float*)d_in[12];  const float* bl2 = (const float*)d_in[13];
    const float* Wl3 = (const float*)d_in[14];  const float* bl3 = (const float*)d_in[15];
    float* out = (float*)d_out;

    // workspace carve-out
    char* w = (char*)d_ws;
    auto alloc = [&](size_t bytes) -> void* {
        void* p = (void*)w;
        w += (bytes + 255) & ~(size_t)255;
        return p;
    };
    float*  dis     = (float*)alloc((size_t)NTOT * 4);
    float*  h       = (float*)alloc((size_t)NTOT * NHID * 4);
    float*  hcat    = (float*)alloc((size_t)NTOT * HCAT * 4);
    float*  hs      = (float*)alloc((size_t)NTOT * 4);
    float*  score   = (float*)alloc((size_t)NTOT * 4);
    float*  readout = (float*)alloc((size_t)BGR * RSIZE * 4);
    float*  z1      = (float*)alloc((size_t)BGR * 256 * 4);
    float*  z2      = (float*)alloc((size_t)BGR * 128 * 4);
    __bf16* xbf     = (__bf16*)alloc((size_t)NTOT * NFEAT * 2);
    __bf16* hb      = (__bf16*)alloc((size_t)NTOT * NHID * 2);
    __bf16* robf    = (__bf16*)alloc((size_t)BGR * RSIZE * 2);
    __bf16* z1bf    = (__bf16*)alloc((size_t)BGR * 256 * 2);
    __bf16* Wt1     = (__bf16*)alloc((size_t)128 * 128 * 2);
    __bf16* Wt2     = (__bf16*)alloc((size_t)128 * 128 * 2);
    __bf16* Wt3     = (__bf16*)alloc((size_t)128 * 128 * 2);
    __bf16* Wtl1    = (__bf16*)alloc((size_t)256 * 768 * 2);
    __bf16* Wtl2    = (__bf16*)alloc((size_t)128 * 256 * 2);

    // degree / dis
    k_init_deg  <<<(NTOT + 255) / 256, 256, 0, stream>>>(dis);
    k_count_deg <<<(ETOT + 255) / 256, 256, 0, stream>>>(dst, dis);
    k_finish_dis<<<(NTOT + 255) / 256, 256, 0, stream>>>(dis);

    // weight / input prep (bf16)
    k_cvt_bf16<<<(NTOT * NFEAT + 255) / 256, 256, 0, stream>>>(x, xbf, NTOT * NFEAT);
    k_transpose_bf16<<<(128 * 128 + 255) / 256, 256, 0, stream>>>(W1,  Wt1,  128, 128);
    k_transpose_bf16<<<(128 * 128 + 255) / 256, 256, 0, stream>>>(W2,  Wt2,  128, 128);
    k_transpose_bf16<<<(128 * 128 + 255) / 256, 256, 0, stream>>>(W3,  Wt3,  128, 128);
    k_transpose_bf16<<<(768 * 256 + 255) / 256, 256, 0, stream>>>(Wl1, Wtl1, 768, 256);
    k_transpose_bf16<<<(256 * 128 + 255) / 256, 256, 0, stream>>>(Wl2, Wtl2, 256, 128);

    const int rowTilesN = NTOT / 16;                    // 4096
    const dim3 gemmGrid(rowTilesN / (8 * 2), 2);        // 8 waves/block, MT=2; 2 col blocks
    const size_t aggLds = (size_t)NPG * NHID * 4;       // 256 KB

    // layer 1
    k_wmma_gemm<4, 128, 2><<<gemmGrid, 256, 0, stream>>>(xbf, NFEAT, Wt1, nullptr,
                                                         h, nullptr, NHID, rowTilesN, 0);
    k_aggregate<<<BGR, 256, aggLds, stream>>>(h, src, dst, dis, b1, hcat, hb, 0);
    // layer 2
    k_wmma_gemm<4, 128, 2><<<gemmGrid, 256, 0, stream>>>(hb, NHID, Wt2, nullptr,
                                                         h, nullptr, NHID, rowTilesN, 0);
    k_aggregate<<<BGR, 256, aggLds, stream>>>(h, src, dst, dis, b2, hcat, hb, NHID);
    // layer 3
    k_wmma_gemm<4, 128, 2><<<gemmGrid, 256, 0, stream>>>(hb, NHID, Wt3, nullptr,
                                                         h, nullptr, NHID, rowTilesN, 0);
    k_aggregate<<<BGR, 256, aggLds, stream>>>(h, src, dst, dis, b3, hcat, hb, 2 * NHID);

    // SAGPool score conv
    k_score_matvec   <<<(NTOT * 32 + 255) / 256, 256, 0, stream>>>(hcat, Ws, hs);
    k_score_aggregate<<<BGR, 256, 0, stream>>>(hs, src, dst, dis, bs, score);
    k_sagpool_readout<<<BGR, 512, 0, stream>>>(hcat, score, readout);

    // MLP head (WMMA with fused bias + relu)
    k_cvt_bf16<<<(BGR * RSIZE + 255) / 256, 256, 0, stream>>>(readout, robf, BGR * RSIZE);
    k_wmma_gemm<16, 768, 1><<<dim3(1, 1), 256, 0, stream>>>(robf, RSIZE, Wtl1, bl1,
                                                            z1, z1bf, 256, BGR / 16, 1);
    k_wmma_gemm<8, 256, 1> <<<dim3(1, 1), 256, 0, stream>>>(z1bf, 256, Wtl2, bl2,
                                                            z2, nullptr, 128, BGR / 16, 1);
    k_logits_lsm<<<1, 128, 0, stream>>>(z2, Wl3, bl3, out);
}
